// GraphAttentionLayer_85624468013734
// MI455X (gfx1250) — matrix operand order repaired and checked by reading, not compiled, not run on hardware
//
#include <hip/hip_runtime.h>
#include <hip/hip_bf16.h>
#include <cstdint>
#include <cstddef>

#define NB   2
#define NN   4096
#define FIN  256
#define NH   8
#define HD   32
#define SLOPE 0.2f

typedef __bf16 bf16_t;
typedef __attribute__((ext_vector_type(8)))  __bf16 v8bf;
typedef __attribute__((ext_vector_type(16))) __bf16 v16bf;
typedef __attribute__((ext_vector_type(8)))  float  v8f;

// ---------------------------------------------------------------------------
// Kernel A: fp32 -> bf16 conversion of h, and W transposed (Wbt[c][f] = W[f][c])
// ---------------------------------------------------------------------------
__global__ void gat_cvt_kernel(const float* __restrict__ hin,
                               const float* __restrict__ Win,
                               bf16_t* __restrict__ hb,
                               bf16_t* __restrict__ Wbt) {
  int idx = blockIdx.x * blockDim.x + threadIdx.x;
  if (idx < NB * NN * FIN) hb[idx] = (bf16_t)hin[idx];
  if (idx < FIN * FIN) {
    int c = idx >> 8, f = idx & 255;          // idx = c*256 + f
    Wbt[idx] = (bf16_t)Win[f * FIN + c];
  }
}

// ---------------------------------------------------------------------------
// Kernel B: Wh = h @ W via v_wmma_f32_16x16x32_bf16.
// One wave per 16x16 output tile; K = 256 in 8 WMMAs.
// Output written transposed: Whbt[b][h][d][n] (bf16), so the attention
// kernel's B-matrix (32x16, K-major) loads are contiguous.
// ---------------------------------------------------------------------------
__global__ void gat_gemm_kernel(const bf16_t* __restrict__ hb,
                                const bf16_t* __restrict__ Wbt,
                                bf16_t* __restrict__ Whbt) {
  int wave = (blockIdx.x * blockDim.x + threadIdx.x) >> 5;
  int lane = threadIdx.x & 31;
  int mt = wave >> 4;                 // 512 row tiles (B*N/16)
  int nt = wave & 15;                 // 16 col tiles (256/16)
  int g0 = mt << 4, c0 = nt << 4;
  int half = lane >> 4, l = lane & 15;

  const bf16_t* arow = hb  + (size_t)(g0 + l) * FIN;   // A row M = l
  const bf16_t* brow = Wbt + (size_t)(c0 + l) * FIN;   // B col N = l (K-major)

  v8f acc = {};
  for (int k = 0; k < FIN; k += 32) {
    // A 16x32 bf16, contiguous k order used here (matches B): lane half selects K block of 16 consecutive
    union { v16bf v; v8bf h2[2]; } A;
    A.h2[0] = *(const v8bf*)(arow + k + half * 16);
    A.h2[1] = *(const v8bf*)(arow + k + half * 16 + 8);
    // B 32x16 bf16, contiguous k order used here (matches A): lane half selects K block of 16 consecutive
    v16bf Bm = *(const v16bf*)(brow + k + half * 16);
    acc = __builtin_amdgcn_wmma_f32_16x16x32_bf16(false, A.v, false, Bm,
                                                  (short)0, acc, false, false);
  }

  // C/D layout: VGPR r -> row M = r + half*8, col N = l
  int c = c0 + l, hh = c >> 5, d = c & 31;
#pragma unroll
  for (int r = 0; r < 8; ++r) {
    int g = g0 + r + half * 8;
    int b = g >> 12, n = g & (NN - 1);
    Whbt[((size_t)(b * NH + hh) * HD + d) * NN + n] = (bf16_t)acc[r];
  }
}

// ---------------------------------------------------------------------------
// Kernel C: attention logit halves  s1[b,h,n] = Wh . a1 ; s2 = Wh . a2
// ---------------------------------------------------------------------------
__global__ void gat_score_kernel(const bf16_t* __restrict__ Whbt,
                                 const float* __restrict__ avec,
                                 float* __restrict__ s1,
                                 float* __restrict__ s2) {
  int t = blockIdx.x * blockDim.x + threadIdx.x;   // (b*NH+h)*NN + n
  int n  = t & (NN - 1);
  int bh = t >> 12;
  int hh = bh & (NH - 1);
  const bf16_t* col = Whbt + (size_t)bh * HD * NN + n;
  float v1 = 0.f, v2 = 0.f;
#pragma unroll
  for (int d = 0; d < HD; ++d) {
    float w = (float)col[(size_t)d * NN];
    v1 += w * avec[hh * 2 * HD + d];
    v2 += w * avec[hh * 2 * HD + HD + d];
  }
  s1[t] = v1;
  s2[t] = v2;
}

// ---------------------------------------------------------------------------
// Kernel D: softmax statistics. One wave per (b,i) computes row max m and
// 1/sum for all 8 heads. adj row read once; mask kept in per-lane bitmask.
// ---------------------------------------------------------------------------
__global__ void gat_stats_kernel(const int* __restrict__ adj,
                                 const float* __restrict__ s1,
                                 const float* __restrict__ s2,
                                 float* __restrict__ mrow,
                                 float* __restrict__ linv) {
  int bi = blockIdx.x;                         // b*NN + i
  int b = bi >> 12, i = bi & (NN - 1);
  int lane = threadIdx.x;
  const int*   adjrow = adj + (size_t)b * NN * NN + (size_t)i * NN;
  const float* s1b = s1 + (size_t)b * NH * NN;
  const float* s2b = s2 + (size_t)b * NH * NN;

  float si[NH], mx[NH], sm[NH];
#pragma unroll
  for (int h = 0; h < NH; ++h) {
    si[h] = s1b[h * NN + i];
    mx[h] = -3.0e38f;
    sm[h] = 0.f;
  }

  unsigned mbits[4] = {0u, 0u, 0u, 0u};        // 128 j's per lane, 1 bit each
  for (int jj = 0; jj < NN / 32; ++jj) {
    int j = jj * 32 + lane;
    if (adjrow[j]) {
      mbits[jj >> 5] |= (1u << (jj & 31));
#pragma unroll
      for (int h = 0; h < NH; ++h) {
        float e = si[h] + s2b[h * NN + j];
        e = e >= 0.f ? e : SLOPE * e;
        mx[h] = fmaxf(mx[h], e);
      }
    }
  }
#pragma unroll
  for (int h = 0; h < NH; ++h)
#pragma unroll
    for (int off = 16; off > 0; off >>= 1)
      mx[h] = fmaxf(mx[h], __shfl_xor(mx[h], off, 32));

  for (int jj = 0; jj < NN / 32; ++jj) {
    if (mbits[jj >> 5] & (1u << (jj & 31))) {
      int j = jj * 32 + lane;
#pragma unroll
      for (int h = 0; h < NH; ++h) {
        float e = si[h] + s2b[h * NN + j];
        e = e >= 0.f ? e : SLOPE * e;
        sm[h] += __expf(e - mx[h]);
      }
    }
  }
#pragma unroll
  for (int h = 0; h < NH; ++h)
#pragma unroll
    for (int off = 16; off > 0; off >>= 1)
      sm[h] += __shfl_xor(sm[h], off, 32);

  if (lane == 0) {
#pragma unroll
    for (int h = 0; h < NH; ++h) {
      mrow[(size_t)(b * NH + h) * NN + i] = mx[h];
      linv[(size_t)(b * NH + h) * NN + i] = 1.f / fmaxf(sm[h], 1e-37f);
    }
  }
}

// ---------------------------------------------------------------------------
// Kernel E: flash-style  out = elu( softmax(e) @ Wh ).
// One wave per (b, h, 16-row tile). Per 32-wide j tile: build bf16 P directly
// in the WMMA A layout, two v_wmma_f32_16x16x32_bf16 (d halves).
// ---------------------------------------------------------------------------
__global__ void gat_attn_kernel(const int* __restrict__ adj,
                                const float* __restrict__ s1,
                                const float* __restrict__ s2,
                                const float* __restrict__ mrow,
                                const float* __restrict__ linv,
                                const bf16_t* __restrict__ Whbt,
                                float* __restrict__ out) {
  int wave = (blockIdx.x * blockDim.x + threadIdx.x) >> 5;
  int lane = threadIdx.x & 31;
  int it = wave & 255;                 // 256 row tiles
  int bh = wave >> 8;
  int b = bh >> 3, hh = bh & 7;
  int half = lane >> 4, l = lane & 15;
  int i0 = it << 4;
  int i  = i0 + l;                     // A-matrix row handled by this lane

  float s1i = s1[(size_t)bh * NN + i];
  float mi  = mrow[(size_t)bh * NN + i];
  float li  = linv[(size_t)bh * NN + i];
  const float* s2b    = s2 + (size_t)bh * NN;
  const int*   adjrow = adj + (size_t)b * NN * NN + (size_t)i * NN;
  const bf16_t* vrow0 = Whbt + ((size_t)bh * HD + l) * NN;        // d = l
  const bf16_t* vrow1 = Whbt + ((size_t)bh * HD + 16 + l) * NN;   // d = 16+l

  int kA = half * 16;       // A K offsets {kA..kA+7} and {kB..kB+7}
  int kB = half * 16 + 8;
  int kV = half * 16;       // B K block of 16 consecutive

  v8f O0 = {}, O1 = {};
  for (int j0 = 0; j0 < NN; j0 += 32) {
    float4 sa0 = *(const float4*)(s2b + j0 + kA);
    float4 sa1 = *(const float4*)(s2b + j0 + kA + 4);
    float4 sb0 = *(const float4*)(s2b + j0 + kB);
    float4 sb1 = *(const float4*)(s2b + j0 + kB + 4);
    int4 aa0 = *(const int4*)(adjrow + j0 + kA);
    int4 aa1 = *(const int4*)(adjrow + j0 + kA + 4);
    int4 ab0 = *(const int4*)(adjrow + j0 + kB);
    int4 ab1 = *(const int4*)(adjrow + j0 + kB + 4);

    float sv[16] = {sa0.x, sa0.y, sa0.z, sa0.w, sa1.x, sa1.y, sa1.z, sa1.w,
                    sb0.x, sb0.y, sb0.z, sb0.w, sb1.x, sb1.y, sb1.z, sb1.w};
    int   av[16] = {aa0.x, aa0.y, aa0.z, aa0.w, aa1.x, aa1.y, aa1.z, aa1.w,
                    ab0.x, ab0.y, ab0.z, ab0.w, ab1.x, ab1.y, ab1.z, ab1.w};

    union { v16bf v; bf16_t e[16]; } A;
#pragma unroll
    for (int q = 0; q < 16; ++q) {
      float e = s1i + sv[q];
      e = e >= 0.f ? e : SLOPE * e;
      float p = av[q] ? __expf(e - mi) * li : 0.f;
      A.e[q] = (bf16_t)p;
    }

    union { v16bf v; v8bf h2[2]; } B0, B1;
    B0.h2[0] = *(const v8bf*)(vrow0 + j0 + kV);
    B0.h2[1] = *(const v8bf*)(vrow0 + j0 + kV + 8);
    B1.h2[0] = *(const v8bf*)(vrow1 + j0 + kV);
    B1.h2[1] = *(const v8bf*)(vrow1 + j0 + kV + 8);

    O0 = __builtin_amdgcn_wmma_f32_16x16x32_bf16(false, A.v, false, B0.v,
                                                 (short)0, O0, false, false);
    O1 = __builtin_amdgcn_wmma_f32_16x16x32_bf16(false, A.v, false, B1.v,
                                                 (short)0, O1, false, false);
  }

  // C layout: VGPR r -> row i0 + r + half*8 ; this lane's columns d=l, d=16+l
#pragma unroll
  for (int r = 0; r < 8; ++r) {
    int iw = i0 + r + half * 8;
    size_t base = ((size_t)b * NN + iw) * (NH * HD) + (size_t)hh * HD;
    float x0 = O0[r];
    x0 = x0 > 0.f ? x0 : __expf(x0) - 1.f;      // ELU (alpha=1)
    out[base + l] = x0;
    float x1 = O1[r];
    x1 = x1 > 0.f ? x1 : __expf(x1) - 1.f;
    out[base + 16 + l] = x1;
  }
}

// ---------------------------------------------------------------------------
extern "C" void kernel_launch(void* const* d_in, const int* in_sizes, int n_in,
                              void* d_out, int out_size, void* d_ws, size_t ws_size,
                              hipStream_t stream) {
  const float* hin  = (const float*)d_in[0];   // (B,N,256) f32
  const int*   adj  = (const int*)d_in[1];     // (B,N,N) i32
  const float* Win  = (const float*)d_in[2];   // (256,256) f32
  const float* avec = (const float*)d_in[3];   // (H,64,1) f32
  float* out = (float*)d_out;                  // (B,N,256) f32

  char* p = (char*)d_ws;
  bf16_t* hb   = (bf16_t*)p; p += (size_t)NB * NN * FIN * 2;       // 4 MB
  bf16_t* Wbt  = (bf16_t*)p; p += (size_t)FIN * FIN * 2;           // 128 KB
  bf16_t* Whbt = (bf16_t*)p; p += (size_t)NB * NH * HD * NN * 2;   // 4 MB
  float* s1    = (float*)p;  p += (size_t)NB * NH * NN * 4;
  float* s2    = (float*)p;  p += (size_t)NB * NH * NN * 4;
  float* mrow  = (float*)p;  p += (size_t)NB * NH * NN * 4;
  float* linv  = (float*)p;  p += (size_t)NB * NH * NN * 4;

  gat_cvt_kernel<<<(NB * NN * FIN + 255) / 256, 256, 0, stream>>>(hin, Win, hb, Wbt);
  gat_gemm_kernel<<<(NB * NN / 16) * (FIN / 16) / 4, 128, 0, stream>>>(hb, Wbt, Whbt);
  gat_score_kernel<<<NB * NH * NN / 256, 256, 0, stream>>>(Whbt, avec, s1, s2);
  gat_stats_kernel<<<NB * NN, 32, 0, stream>>>(adj, s1, s2, mrow, linv);
  gat_attn_kernel<<<NB * NH * (NN / 16) / 4, 128, 0, stream>>>(adj, s1, s2, mrow,
                                                               linv, Whbt, out);
}
